// SimpleHyenaBlock_13022340841882
// MI455X (gfx1250) — matrix operand order repaired and checked by reading, not compiled, not run on hardware
//
#include <hip/hip_runtime.h>
#include <hip/hip_bf16.h>

// ---------------------------------------------------------------------------
// SimpleHyenaBlock on MI455X (gfx1250), wave32 + WMMA bf16 + async-to-LDS.
//   x_proj = x @ in_w + in_b
//   h      = gelu(x @ mod_w1 + mod_b1)           (exact gelu)
//   mod    = h @ mod_w2 + mod_b2
//   m      = x_proj * mod
//   y[t,d] = sum_{k<64} kernel[d,k] * m[(t-k) mod L, d]   (FFT conv == 64-tap circular FIR)
//   out    = y @ out_w + out_b
// All weights are pre-transposed once to Wt[N][K] bf16 so both GEMM operands
// stream into LDS as contiguous 16B chunks via GLOBAL_LOAD_ASYNC_TO_LDS_B128.
// ---------------------------------------------------------------------------

typedef __attribute__((ext_vector_type(16))) __bf16 v16bf;
typedef __attribute__((ext_vector_type(8)))  float  v8f;
typedef int v4i_ __attribute__((__vector_size__(16)));   // exact builtin pointee

#define DIMC  1024
#define HIDC  4096
#define BSZ   4
#define LSEQ  4096
#define MROWS (BSZ * LSEQ)   // 16384
#define KTAPS 64

#if defined(__has_builtin)
#if __has_builtin(__builtin_amdgcn_global_load_async_to_lds_b128) && \
    __has_builtin(__builtin_amdgcn_s_wait_asynccnt)
#define HAS_ASYNC_LDS 1
#endif
#endif
#ifndef HAS_ASYNC_LDS
#define HAS_ASYNC_LDS 0
#endif

#define GLB_V4(p) ((__attribute__((address_space(1))) v4i_*)(p))
#define LDS_V4(p) ((__attribute__((address_space(3))) v4i_*)(p))

static __device__ __forceinline__ __bf16 f2bf(float f) {
  union { float f; unsigned u; } x; x.f = f;
  unsigned r = (x.u + 0x7FFFu + ((x.u >> 16) & 1u)) >> 16;
  unsigned short hu = (unsigned short)r;
  __bf16 h;
  __builtin_memcpy(&h, &hu, 2);
  return h;
}

// ---------------------------------------------------------------------------
// fp32 -> bf16 elementwise convert (grid-stride)
// ---------------------------------------------------------------------------
__global__ void cvt_f32_bf16(const float* __restrict__ in,
                             __bf16* __restrict__ out, int n) {
  for (int i = blockIdx.x * blockDim.x + threadIdx.x; i < n;
       i += gridDim.x * blockDim.x) {
    out[i] = f2bf(in[i]);
  }
}

// ---------------------------------------------------------------------------
// fp32 [K][N] -> bf16 transposed [N][K]   (one-time weight prep, 32x32 tiles)
// ---------------------------------------------------------------------------
__global__ __launch_bounds__(256)
void cvt_transpose_bf16(const float* __restrict__ in, __bf16* __restrict__ out,
                        int K, int N) {
  __shared__ float tile[32][33];
  const int k0 = blockIdx.y * 32;
  const int n0 = blockIdx.x * 32;
  const int tx = threadIdx.x & 31;
  const int ty = threadIdx.x >> 5;          // 0..7
#pragma unroll
  for (int r = ty; r < 32; r += 8)
    tile[r][tx] = in[(size_t)(k0 + r) * N + n0 + tx];
  __syncthreads();
#pragma unroll
  for (int r = ty; r < 32; r += 8)
    out[(size_t)(n0 + r) * K + k0 + tx] = f2bf(tile[tx][r]);
}

// ---------------------------------------------------------------------------
// Tiled bf16 WMMA GEMM:  C[M,N] = A[M,K] * Bt[N,K]^T + bias, fused epilogue.
//   mode 0: outF = acc + bias
//   mode 1: outH = bf16(gelu(acc + bias))
//   mode 2: outF = aux * (acc + bias)        (gating)
// Block: 256 threads = 8 waves. Tile BM=64, BN=128, BK=32, double-buffered
// LDS staged with GLOBAL_LOAD_ASYNC_TO_LDS_B128 (ASYNCcnt), 3 chunks/thread.
// Wave (wm = wid&3, wn = wid>>2) computes rows [wm*16,+16) x cols [wn*64,+64).
// ---------------------------------------------------------------------------
#define BM 64
#define BN 128
#define BK 32
#define LDT 40   // padded K-stride of LDS tiles (80B rows, 16B aligned)

#if HAS_ASYNC_LDS
static __device__ __forceinline__ void stage_async(
    const __bf16* __restrict__ Ag,   // block A base:  A  + m0*K (+kbase below)
    const __bf16* __restrict__ Bg,   // block Bt base: Bt + n0*K
    int K, int kbase,
    __bf16 (*As)[LDT], __bf16 (*Bs)[LDT], int tid) {
  // A tile: 64 rows x 32 k = 256 x 16B chunks, one per thread
  {
    const int row = tid >> 2;
    const int kc  = (tid & 3) * 8;
    __builtin_amdgcn_global_load_async_to_lds_b128(
        GLB_V4(Ag + (size_t)row * K + kbase + kc),
        LDS_V4(&As[row][kc]), 0, 0);
  }
  // Bt tile: 128 rows x 32 k = 512 x 16B chunks, two per thread
#pragma unroll
  for (int p = 0; p < 2; ++p) {
    const int c   = tid + p * 256;
    const int row = c >> 2;
    const int kc  = (c & 3) * 8;
    __builtin_amdgcn_global_load_async_to_lds_b128(
        GLB_V4(Bg + (size_t)row * K + kbase + kc),
        LDS_V4(&Bs[row][kc]), 0, 0);
  }
}
#else
static __device__ __forceinline__ void stage_sync(
    const __bf16* __restrict__ Ag, const __bf16* __restrict__ Bg,
    int K, int kbase,
    __bf16 (*As)[LDT], __bf16 (*Bs)[LDT], int tid) {
  const int arow = tid >> 2;
  const int akc  = (tid & 3) * 8;
  const uint4 qa = *(const uint4*)(Ag + (size_t)arow * K + kbase + akc);
  uint4 qb[2];
#pragma unroll
  for (int p = 0; p < 2; ++p) {
    const int c = tid + p * 256;
    qb[p] = *(const uint4*)(Bg + (size_t)(c >> 2) * K + kbase + (c & 3) * 8);
  }
  *(uint4*)(&As[arow][akc]) = qa;
#pragma unroll
  for (int p = 0; p < 2; ++p) {
    const int c = tid + p * 256;
    *(uint4*)(&Bs[c >> 2][(c & 3) * 8]) = qb[p];
  }
}
#endif

__global__ __launch_bounds__(256)
void gemm_bf16_wmma(const __bf16* __restrict__ A,    // [M][K]
                    const __bf16* __restrict__ Bt,   // [N][K] (transposed)
                    const float*  __restrict__ bias,
                    const float*  __restrict__ aux,
                    float*        __restrict__ outF,
                    __bf16*       __restrict__ outH,
                    int M, int N, int K, int mode) {
  __shared__ alignas(16) __bf16 As[2][BM][LDT];
  __shared__ alignas(16) __bf16 Bs[2][BN][LDT];

  const int tid  = threadIdx.x;
  const int lane = tid & 31;
  const int wid  = tid >> 5;
  const int wm   = wid & 3;        // 0..3 -> M sub-tile
  const int wn   = wid >> 2;       // 0..1 -> N half
  const int laneHi = lane >> 4;    // 0/1
  const int laneLo = lane & 15;

  const int m0 = blockIdx.y * BM;
  const int n0 = blockIdx.x * BN;
  const __bf16* Ag = A  + (size_t)m0 * K;
  const __bf16* Bg = Bt + (size_t)n0 * K;

  v8f acc[4];
#pragma unroll
  for (int s = 0; s < 4; ++s)
#pragma unroll
    for (int r = 0; r < 8; ++r) acc[s][r] = 0.0f;

  const int nk = K / BK;

#if HAS_ASYNC_LDS
  stage_async(Ag, Bg, K, 0, As[0], Bs[0], tid);
#else
  stage_sync(Ag, Bg, K, 0, As[0], Bs[0], tid);
#endif

  for (int kt = 0; kt < nk; ++kt) {
    const int cur = kt & 1;
#if HAS_ASYNC_LDS
    if (kt + 1 < nk) {
      stage_async(Ag, Bg, K, (kt + 1) * BK, As[1 - cur], Bs[1 - cur], tid);
      __builtin_amdgcn_s_wait_asynccnt(3);   // tile kt's 3 copies are done
    } else {
      __builtin_amdgcn_s_wait_asynccnt(0);
    }
#else
    if (kt + 1 < nk) {
      const int row = tid >> 2;
      __builtin_prefetch(Ag + (size_t)row * K + (kt + 1) * BK, 0, 1);
      __builtin_prefetch(Bg + (size_t)(tid >> 1) * K + (kt + 1) * BK, 0, 1);
    }
#endif
    __syncthreads();   // all waves' copies for tile kt landed

    // ---- A fragment (16x32 bf16): two contiguous 16B chunks per lane ------
    // lanes 0-15: M=lane,    K = 0..7  and 16..23
    // lanes16-31: M=lane-16, K = 8..15 and 24..31
    union { uint4 q[2]; v16bf v; } af;
    {
      const int ar = wm * 16 + laneLo;
      af.q[0] = *(const uint4*)(&As[cur][ar][laneHi * 8]);
      af.q[1] = *(const uint4*)(&As[cur][ar][16 + laneHi * 8]);
    }

    // ---- 4 N sub-tiles: B fragment (N=laneLo, K same as A fragment) + WMMA ---
#pragma unroll
    for (int s = 0; s < 4; ++s) {
      union { uint4 q[2]; v16bf v; } bf;
      const int bn = wn * 64 + s * 16 + laneLo;
      bf.q[0] = *(const uint4*)(&Bs[cur][bn][laneHi * 8]);
      bf.q[1] = *(const uint4*)(&Bs[cur][bn][16 + laneHi * 8]);
      acc[s] = __builtin_amdgcn_wmma_f32_16x16x32_bf16(
          /*neg_a=*/false, af.v, /*neg_b=*/false, bf.v,
          /*c_mod=*/(short)0, acc[s], /*reuse_a=*/false, /*reuse_b=*/false);
    }
    __syncthreads();   // done reading buf[cur]; next iter may overwrite it

#if !HAS_ASYNC_LDS
    if (kt + 1 < nk)
      stage_sync(Ag, Bg, K, (kt + 1) * BK, As[1 - cur], Bs[1 - cur], tid);
#endif
  }

  // ---- epilogue: C/D layout (VGPR r -> M = r + 8*laneHi, N = laneLo) -------
#pragma unroll
  for (int s = 0; s < 4; ++s) {
    const int nglob = n0 + wn * 64 + s * 16 + laneLo;
    const float bv = bias[nglob];
#pragma unroll
    for (int r = 0; r < 8; ++r) {
      const int mglob = m0 + wm * 16 + r + 8 * laneHi;
      const size_t idx = (size_t)mglob * N + nglob;
      float v = acc[s][r] + bv;
      if (mode == 0) {
        outF[idx] = v;
      } else if (mode == 1) {
        float g = 0.5f * v * (1.0f + erff(v * 0.70710678118654752f));
        outH[idx] = f2bf(g);
      } else {
        outF[idx] = aux[idx] * v;
      }
    }
  }
}

// ---------------------------------------------------------------------------
// 64-tap circular FIR over the sequence axis, per channel.
//   y[b,t,d] = sum_k kern[d,k] * m[b,(t-k) mod L, d], output as bf16.
// Tile: 64 t x 64 d per block; halo of 63 t rows with mod-L wrap.
// ---------------------------------------------------------------------------
__global__ __launch_bounds__(256)
void fir_conv64(const float* __restrict__ modv,   // [B*L, D]
                const float* __restrict__ kern,   // [D, 64]
                __bf16* __restrict__ y) {         // [B*L, D] bf16
  __shared__ float ws[64][65];        // [channel][tap], conflict-free
  __shared__ float sm[127][64];       // [t-local][channel]

  const int tid = threadIdx.x;
  const int t0  = blockIdx.x * 64;
  const int d0  = blockIdx.y * 64;
  const int b   = blockIdx.z;

  for (int idx = tid; idx < 64 * 64; idx += 256) {
    const int c = idx >> 6, k = idx & 63;
    ws[c][k] = kern[(size_t)(d0 + c) * KTAPS + k];
  }
  for (int idx = tid; idx < 127 * 64; idx += 256) {
    const int row = idx >> 6, c = idx & 63;
    int t = t0 - 63 + row;
    t = (t + LSEQ) % LSEQ;
    sm[row][c] = modv[((size_t)b * LSEQ + t) * DIMC + d0 + c];
  }
  __syncthreads();

  const int c  = tid & 63;
  const int tg = tid >> 6;            // 0..3, each thread does 16 t values
#pragma unroll
  for (int j = 0; j < 16; ++j) {
    const int tl = tg * 16 + j;
    float s = 0.0f;
#pragma unroll
    for (int k = 0; k < KTAPS; ++k) {
      s += ws[c][k] * sm[tl + 63 - k][c];
    }
    y[((size_t)b * LSEQ + t0 + tl) * DIMC + d0 + c] = f2bf(s);
  }
}

// ---------------------------------------------------------------------------
// launch
// ---------------------------------------------------------------------------
extern "C" void kernel_launch(void* const* d_in, const int* in_sizes, int n_in,
                              void* d_out, int out_size, void* d_ws, size_t ws_size,
                              hipStream_t stream) {
  const float* x      = (const float*)d_in[0];   // [B,L,D]
  const float* in_w   = (const float*)d_in[1];   // [D,D]
  const float* in_b   = (const float*)d_in[2];   // [D]
  const float* mod_w1 = (const float*)d_in[3];   // [D,H]
  const float* mod_b1 = (const float*)d_in[4];   // [H]
  const float* mod_w2 = (const float*)d_in[5];   // [H,D]
  const float* mod_b2 = (const float*)d_in[6];   // [D]
  const float* kern   = (const float*)d_in[7];   // [D,64]
  const float* out_w  = (const float*)d_in[8];   // [D,D]
  const float* out_b  = (const float*)d_in[9];   // [D]
  float* out = (float*)d_out;                    // [B,L,D]

  char* w = (char*)d_ws;
  __bf16* x_bf   = (__bf16*)(w);                           // 32 MB
  __bf16* wt_in  = (__bf16*)(w + 33554432);                //  2 MB  [D][D]^T
  __bf16* wt_m1  = (__bf16*)(w + 35651584);                //  8 MB  [H][D]
  __bf16* wt_m2  = (__bf16*)(w + 44040192);                //  8 MB  [D][H]
  __bf16* wt_out = (__bf16*)(w + 52428800);                //  2 MB  [D][D]^T
  float*  x_proj = (float*)(w + 54525952);                 // 64 MB
  __bf16* h_bf   = (__bf16*)(w + 121634816);               // 128 MB
  float*  modv   = (float*)(w + 255852544);                // 64 MB
  __bf16* y_bf   = (__bf16*)(w + 322961408);               // 32 MB
  (void)ws_size; (void)in_sizes; (void)n_in; (void)out_size;

  // 1) activations fp32 -> bf16; weights fp32 [K][N] -> bf16 transposed [N][K]
  cvt_f32_bf16<<<4096, 256, 0, stream>>>(x, x_bf, MROWS * DIMC);
  cvt_transpose_bf16<<<dim3(DIMC / 32, DIMC / 32), 256, 0, stream>>>(in_w,   wt_in,  DIMC, DIMC);
  cvt_transpose_bf16<<<dim3(HIDC / 32, DIMC / 32), 256, 0, stream>>>(mod_w1, wt_m1,  DIMC, HIDC);
  cvt_transpose_bf16<<<dim3(DIMC / 32, HIDC / 32), 256, 0, stream>>>(mod_w2, wt_m2,  HIDC, DIMC);
  cvt_transpose_bf16<<<dim3(DIMC / 32, DIMC / 32), 256, 0, stream>>>(out_w,  wt_out, DIMC, DIMC);

  // 2) x_proj = x @ in_w + in_b                       [16384,1024]
  gemm_bf16_wmma<<<dim3(DIMC / BN, MROWS / BM), 256, 0, stream>>>(
      x_bf, wt_in, in_b, nullptr, x_proj, nullptr, MROWS, DIMC, DIMC, 0);

  // 3) h = gelu(x @ mod_w1 + mod_b1) -> bf16          [16384,4096]
  gemm_bf16_wmma<<<dim3(HIDC / BN, MROWS / BM), 256, 0, stream>>>(
      x_bf, wt_m1, mod_b1, nullptr, nullptr, h_bf, MROWS, HIDC, DIMC, 1);

  // 4) modulated = x_proj * (h @ mod_w2 + mod_b2)     [16384,1024]
  gemm_bf16_wmma<<<dim3(DIMC / BN, MROWS / BM), 256, 0, stream>>>(
      h_bf, wt_m2, mod_b2, x_proj, modv, nullptr, MROWS, DIMC, HIDC, 2);

  // 5) y = circular FIR(modulated, kernel) -> bf16    [16384,1024]
  fir_conv64<<<dim3(LSEQ / 64, DIMC / 64, BSZ), 256, 0, stream>>>(
      modv, kern, y_bf);

  // 6) out = y @ out_w + out_b                        [16384,1024]
  gemm_bf16_wmma<<<dim3(DIMC / BN, MROWS / BM), 256, 0, stream>>>(
      y_bf, wt_out, out_b, nullptr, out, nullptr, MROWS, DIMC, DIMC, 0);
}
